// Attention_41283225649336
// MI455X (gfx1250) — matrix operand order repaired and checked by reading, not compiled, not run on hardware
//
#include <hip/hip_runtime.h>

#define DIM      1024
#define NSEQ     2048
#define BATCH    2
#define HEADS    16
#define HEAD_DIM 64

typedef __attribute__((ext_vector_type(16))) __bf16          v16bf;
typedef __attribute__((ext_vector_type(8)))  float           v8f;
typedef __attribute__((ext_vector_type(8)))  unsigned short  us8;
typedef __attribute__((ext_vector_type(16))) unsigned short  us16;
typedef __attribute__((ext_vector_type(4)))  unsigned int    v4u;
typedef __attribute__((ext_vector_type(8)))  int             v8i;
typedef __attribute__((ext_vector_type(4)))  int             v4i;

// fp32 -> bf16 (RNE) via native cast; backend emits v_cvt_*bf16_f32
static __device__ __forceinline__ unsigned short f2bf(float f) {
  __bf16 h = (__bf16)f;
  return __builtin_bit_cast(unsigned short, h);
}

// A-matrix 16x32 bf16 fragment: lane holds row M=lane%16;
// K = 2*off..2*off+15 contiguous, off = (lane>=16)*8.
static __device__ __forceinline__ v16bf load_a16x32(const unsigned short* rowp, int off) {
  us8 lo = *(const us8*)(rowp + 2 * off);
  us8 hi = *(const us8*)(rowp + 2 * off + 8);
  us16 c = __builtin_shufflevector(lo, hi, 0,1,2,3,4,5,6,7,8,9,10,11,12,13,14,15);
  return __builtin_bit_cast(v16bf, c);
}
// B-matrix 32x16 bf16 fragment from LDS stored [N][K]: lane holds column
// N=lane%16, K = koff..koff+15 contiguous (koff = (lane>=16)*16 + chunk base).
static __device__ __forceinline__ v16bf load_b32x16(const unsigned short* colp, int koff) {
  us8 lo = *(const us8*)(colp + koff);
  us8 hi = *(const us8*)(colp + koff + 8);
  us16 c = __builtin_shufflevector(lo, hi, 0,1,2,3,4,5,6,7,8,9,10,11,12,13,14,15);
  return __builtin_bit_cast(v16bf, c);
}
static __device__ __forceinline__ v8f wmma_bf16(v16bf a, v16bf b, v8f c) {
  return __builtin_amdgcn_wmma_f32_16x16x32_bf16(false, a, false, b, (short)0, c, false, false);
}

// Async global->LDS 16B copy (CDNA5, ASYNCcnt tracked). Per-lane addresses;
// one wave-wide instruction moves 512B. GVS mode: saddr base + vgpr u32 offset.
static __device__ __forceinline__ void async_copy16(const void* base, unsigned goff, void* ldsdst) {
  unsigned laddr = (unsigned)(size_t)ldsdst;  // low 32 bits of generic ptr = LDS offset
  asm volatile("global_load_async_to_lds_b128 %0, %1, %2"
               :: "v"(laddr), "v"(goff), "s"(base) : "memory");
}

// Tensor Data Mover: 2D tile Global->LDS per D# (CDNA5 ISA).
// dim0 = contiguous axis (elements of 2B); LDS padding injected per row so the
// landed tile has a bank-friendly stride. Wave-level op, TENSORcnt tracked.
// This toolchain exposes the 6-arg builtin: (g0, g1, g2, g3, g_extra, cpol).
static __device__ __forceinline__ void tdm_load_2d(
    const void* gaddr, void* ldsdst,
    unsigned tile_d0, unsigned tile_d1,       // tile dims (elements)
    unsigned tensor_d0, unsigned tensor_d1,   // tensor dims (OOB bounds)
    unsigned d0_stride,                       // elements between dim1 lines
    unsigned g1w0)                            // group1 dword0: data_size + pad cfg
{
  unsigned long long ga = (unsigned long long)(size_t)gaddr;
  v4u g0;
  g0[0] = 1u;                                  // count=1, user descriptor
  g0[1] = (unsigned)(size_t)ldsdst;            // lds_addr (bytes)
  g0[2] = (unsigned)ga;                        // global_addr[31:0]
  g0[3] = (unsigned)((ga >> 32) & 0x1FFFFFFu) | (2u << 30);  // addr[56:32] | type=2
  v8i g1;
  g1[0] = (int)g1w0;                           // wg_mask=0 | data_size | pad cfg
  g1[1] = (int)((tensor_d0 & 0xFFFFu) << 16);  // atomic_barrier=0 | tensor_dim0 lo16
  g1[2] = (int)(((tensor_d0 >> 16) & 0xFFFFu) | ((tensor_d1 & 0xFFFFu) << 16));
  g1[3] = (int)(((tensor_d1 >> 16) & 0xFFFFu) | (tile_d0 << 16));  // | tile_dim0
  g1[4] = (int)tile_d1;                        // tile_dim1 | tile_dim2=0
  g1[5] = (int)d0_stride;                      // tensor_dim0_stride[31:0]
  g1[6] = 0;                                   // stride hi | dim1_stride lo (2D: unused)
  g1[7] = 0;
  v4i z4 = {0, 0, 0, 0};
  v8i z8 = {0, 0, 0, 0, 0, 0, 0, 0};
  __builtin_amdgcn_tensor_load_to_lds(g0, g1, z4, z4, z8, 0);
}
// group1 dword0 configs: data_size=2B (1<<16), pad_enable (1<<20),
// pad_interval code<<22, pad_amount code<<25
#define TDM_G1_K (0x10000u | 0x100000u | (4u << 22) | (3u << 25))  // 32DW rows +4DW -> stride 72
#define TDM_G1_V (0x10000u | 0x100000u | (3u << 22) | (3u << 25))  // 16DW rows +4DW -> stride 40

#define LDK 40  // LDS K-stride (elements): 80 bytes, 16B-aligned, bank-spreading

// ---------------------------------------------------------------------------
// Kernel 1: QKV = x @ w_qkv, fp32 in, bf16 out. Q: [b,h,n,64] pre-scaled by
// 1/sqrt(64). K: [b,h,n,64]. V stored TRANSPOSED [b,h,64,n] so the attention
// kernel can TDM-copy V^T tiles without a transpose.
// ---------------------------------------------------------------------------
__global__ void __launch_bounds__(256)
qkv_gemm(const float* __restrict__ x, const float* __restrict__ w,
         unsigned short* __restrict__ qb, unsigned short* __restrict__ kbuf,
         unsigned short* __restrict__ vbuf) {
  __shared__ unsigned short ldsA[128 * LDK];   // [m][k]
  __shared__ unsigned short ldsBt[128 * LDK];  // [n][k] (transposed)
  const int tid = threadIdx.x;
  const int lane = tid & 31, wid = tid >> 5;
  const int lane16 = lane & 15, hf = lane >> 4;
  const int wm = wid >> 2, wn = wid & 3;       // 2x4 wave grid, 64x32 per wave
  const int m0 = blockIdx.y * 128, n0 = blockIdx.x * 128;

  v8f acc[4][2] = {};
  for (int ks = 0; ks < DIM / 32; ++ks) {
    // stage A tile 128x32 (fp32 -> bf16)
#pragma unroll
    for (int it = 0; it < 4; ++it) {
      int slot = tid + it * 256;
      int r = slot >> 3, kq = (slot & 7) * 4;
      float4 f = *(const float4*)(x + (size_t)(m0 + r) * DIM + ks * 32 + kq);
      unsigned short* d = &ldsA[r * LDK + kq];
      d[0] = f2bf(f.x); d[1] = f2bf(f.y); d[2] = f2bf(f.z); d[3] = f2bf(f.w);
    }
    // stage B tile 32x128 transposed -> [n][k]
#pragma unroll
    for (int it = 0; it < 4; ++it) {
      int slot = tid + it * 256;
      int kr = slot >> 5, nq = (slot & 31) * 4;
      float4 f = *(const float4*)(w + (size_t)(ks * 32 + kr) * 3072 + n0 + nq);
      ldsBt[(nq + 0) * LDK + kr] = f2bf(f.x);
      ldsBt[(nq + 1) * LDK + kr] = f2bf(f.y);
      ldsBt[(nq + 2) * LDK + kr] = f2bf(f.z);
      ldsBt[(nq + 3) * LDK + kr] = f2bf(f.w);
    }
    __syncthreads();
    v16bf af[4], bfr[2];
#pragma unroll
    for (int mf = 0; mf < 4; ++mf)
      af[mf] = load_a16x32(&ldsA[(wm * 64 + mf * 16 + lane16) * LDK], hf * 8);
#pragma unroll
    for (int nf = 0; nf < 2; ++nf)
      bfr[nf] = load_b32x16(&ldsBt[(wn * 32 + nf * 16 + lane16) * LDK], hf * 16);
#pragma unroll
    for (int mf = 0; mf < 4; ++mf)
#pragma unroll
      for (int nf = 0; nf < 2; ++nf)
        acc[mf][nf] = wmma_bf16(af[mf], bfr[nf], acc[mf][nf]);
    __syncthreads();
  }
  // epilogue: section id is uniform per 16-col fragment -> scalar branch
#pragma unroll
  for (int nf = 0; nf < 2; ++nf) {
    const int gcb = n0 + wn * 32 + nf * 16;  // block/wave-uniform
    const int sec = gcb >> 10;               // 0=Q 1=K 2=V (uniform)
    const int inner = gcb & 1023;
    const int h = inner >> 6;
    const int d = (inner & 63) + lane16;     // head dim index for this lane
#pragma unroll
    for (int mf = 0; mf < 4; ++mf)
#pragma unroll
      for (int i = 0; i < 8; ++i) {
        int gm = m0 + wm * 64 + mf * 16 + i + hf * 8;   // row in [0,4096)
        int bb = gm >> 11, nn = gm & 2047;
        int bh = bb * HEADS + h;
        float v = acc[mf][nf][i];
        if (sec == 0)
          qb[((size_t)bh * NSEQ + nn) * HEAD_DIM + d] = f2bf(v * 0.125f);
        else if (sec == 1)
          kbuf[((size_t)bh * NSEQ + nn) * HEAD_DIM + d] = f2bf(v);
        else
          vbuf[((size_t)bh * HEAD_DIM + d) * NSEQ + nn] = f2bf(v);  // V^T
      }
  }
}

// ---------------------------------------------------------------------------
// Kernel 2: flash attention. Grid (N/64, B*H); 4 waves, 16 Q-rows per wave.
// K[j][d] and V^T[d][j] tiles staged by the Tensor Data Mover (wave 0 issues
// two D# per tile; TENSORcnt double-buffered). TDM pad config lands the tiles
// at bank-friendly LDS strides 72/40 directly.
// scores = Q@K^T (4 wmma), online softmax, O += P@V (4 wmma).
// ---------------------------------------------------------------------------
#define LDD 72  // k-tile row stride ([j][d], d=64)  == 32DW + 4DW TDM pad
#define LDJ 40  // v^T row stride   ([d][j], j=32)  == 16DW + 4DW TDM pad
#define LDP 40  // P row stride     ([m][j], j=32)
#define NJT (NSEQ / 32)
__global__ void __launch_bounds__(128)
attn_kernel(const unsigned short* __restrict__ qb, const unsigned short* __restrict__ kb,
            const unsigned short* __restrict__ vbt, unsigned short* __restrict__ attnb) {
  __shared__ unsigned short ldsK[2][32 * LDD];   // K tile [j][d] == B-layout [N][K]
  __shared__ unsigned short ldsVt[2][64 * LDJ];  // V^T tile [d][j]
  __shared__ unsigned short ldsP[4][16 * LDP];   // per-wave P relayout C->A
  const int tid = threadIdx.x;
  const int lane = tid & 31, wid = tid >> 5;
  const int lane16 = lane & 15, hf = lane >> 4;
  const int bh = blockIdx.y;
  const int m_base = blockIdx.x * 64 + wid * 16;
  const unsigned short* kg = kb + (size_t)bh * NSEQ * HEAD_DIM;   // [n][d]
  const unsigned short* vg = vbt + (size_t)bh * HEAD_DIM * NSEQ;  // [d][n]

  // persistent Q A-fragments (16 rows x 64 d = 2 frags of 16x32)
  const unsigned short* qrow = qb + ((size_t)bh * NSEQ + m_base + lane16) * HEAD_DIM;
  v16bf qf0 = load_a16x32(qrow, hf * 8);
  v16bf qf1 = load_a16x32(qrow + 32, hf * 8);

  // TDM-stage one 32-key tile: two descriptors (K tile, V^T tile)
  auto stage = [&](int jt, int buf) {
    int j0 = jt * 32;
    // K: dim0 = d (64 contig), 32 rows of stride HEAD_DIM
    tdm_load_2d(kg + (size_t)j0 * HEAD_DIM, &ldsK[buf][0],
                /*tile*/ HEAD_DIM, 32, /*tensor*/ HEAD_DIM, NSEQ,
                /*stride*/ HEAD_DIM, TDM_G1_K);
    // V^T: dim0 = j (32 contig), 64 rows of stride NSEQ
    tdm_load_2d(vg + j0, &ldsVt[buf][0],
                /*tile*/ 32, HEAD_DIM, /*tensor*/ NSEQ, HEAD_DIM,
                /*stride*/ NSEQ, TDM_G1_V);
  };

  float mrun[8], srun[8];
#pragma unroll
  for (int i = 0; i < 8; ++i) { mrun[i] = -1e30f; srun[i] = 0.f; }
  v8f o_[4] = {};  // O accumulator 16x64 fp32

  if (wid == 0) stage(0, 0);
  for (int jt = 0; jt < NJT; ++jt) {
    const int buf = jt & 1;
    if (wid == 0) {  // wave-uniform branch; TDM ignores EXEC, one wave issues
      if (jt + 1 < NJT) {
        stage(jt + 1, buf ^ 1);                      // prefetch next tile
        __builtin_amdgcn_s_wait_tensorcnt(2);        // tile jt landed (in-order)
      } else {
        __builtin_amdgcn_s_wait_tensorcnt(0);
      }
    }
    __syncthreads();  // TDM data visible to all waves

    // scores 16x32: K tile already in B layout [j][d]
    v8f sc0 = {}, sc1 = {};
    {
      const unsigned short* c0 = &ldsK[buf][lane16 * LDD];
      const unsigned short* c1 = &ldsK[buf][(16 + lane16) * LDD];
      sc0 = wmma_bf16(qf0, load_b32x16(c0, hf * 16), sc0);
      sc0 = wmma_bf16(qf1, load_b32x16(c0 + 32, hf * 16), sc0);
      sc1 = wmma_bf16(qf0, load_b32x16(c1, hf * 16), sc1);
      sc1 = wmma_bf16(qf1, load_b32x16(c1 + 32, hf * 16), sc1);
    }

    // online softmax: C-layout row = i + hf*8, spread over 16 lanes
    unsigned short* pw = &ldsP[wid][0];
#pragma unroll
    for (int i = 0; i < 8; ++i) {
      float a0 = sc0[i], a1 = sc1[i];
      float t = fmaxf(a0, a1);
      t = fmaxf(t, __shfl_xor(t, 1, 16));
      t = fmaxf(t, __shfl_xor(t, 2, 16));
      t = fmaxf(t, __shfl_xor(t, 4, 16));
      t = fmaxf(t, __shfl_xor(t, 8, 16));
      float mnew = fmaxf(mrun[i], t);
      float p0 = __expf(a0 - mnew);
      float p1 = __expf(a1 - mnew);
      float ps = p0 + p1;
      ps += __shfl_xor(ps, 1, 16);
      ps += __shfl_xor(ps, 2, 16);
      ps += __shfl_xor(ps, 4, 16);
      ps += __shfl_xor(ps, 8, 16);
      float alpha = __expf(mrun[i] - mnew);
      srun[i] = srun[i] * alpha + ps;
      mrun[i] = mnew;
#pragma unroll
      for (int df = 0; df < 4; ++df) o_[df][i] *= alpha;
      pw[(i + hf * 8) * LDP + lane16]      = f2bf(p0);
      pw[(i + hf * 8) * LDP + 16 + lane16] = f2bf(p1);
    }
    // wave-local LDS RAW: DS ops are in-order per wave; fence the compiler + hw
    asm volatile("s_wait_dscnt 0" ::: "memory");
    v16bf pf = load_a16x32(&pw[lane16 * LDP], hf * 8);
#pragma unroll
    for (int df = 0; df < 4; ++df) {
      v16bf vbfr = load_b32x16(&ldsVt[buf][(df * 16 + lane16) * LDJ], hf * 16);
      o_[df] = wmma_bf16(pf, vbfr, o_[df]);
    }
    __syncthreads();  // done reading buf before it is overwritten at jt+2
  }

  const int b = bh >> 4, h = bh & 15;
#pragma unroll
  for (int i = 0; i < 8; ++i) {
    float inv = 1.0f / srun[i];
    int row = m_base + i + hf * 8;
    size_t base = ((size_t)b * NSEQ + row) * DIM + h * HEAD_DIM;
#pragma unroll
    for (int df = 0; df < 4; ++df)
      attnb[base + df * 16 + lane16] = f2bf(o_[df][i] * inv);
  }
}

// ---------------------------------------------------------------------------
// Kernel 3: out = attn(bf16) @ w_out(fp32->bf16) + b_out, fp32 result.
// A tile staged with async global->LDS copies (already bf16).
// ---------------------------------------------------------------------------
__global__ void __launch_bounds__(256)
out_gemm(const unsigned short* __restrict__ a, const float* __restrict__ w,
         const float* __restrict__ bias, float* __restrict__ out) {
  __shared__ unsigned short ldsA[128 * LDK];
  __shared__ unsigned short ldsBt[128 * LDK];
  const int tid = threadIdx.x;
  const int lane = tid & 31, wid = tid >> 5;
  const int lane16 = lane & 15, hf = lane >> 4;
  const int wm = wid >> 2, wn = wid & 3;
  const int m0 = blockIdx.y * 128, n0 = blockIdx.x * 128;

  v8f acc[4][2] = {};
  for (int ks = 0; ks < DIM / 32; ++ks) {
    // A tile (bf16): async copy, no VGPR round-trip
#pragma unroll
    for (int it = 0; it < 2; ++it) {
      int c = tid + it * 256;
      int r = c >> 2, kq = (c & 3) * 8;
      async_copy16(a, (unsigned)(((unsigned)(m0 + r) * DIM + ks * 32 + kq) * 2),
                   &ldsA[r * LDK + kq]);
    }
    // B tile 32x128 fp32 -> bf16, transposed -> [n][k]
#pragma unroll
    for (int it = 0; it < 4; ++it) {
      int slot = tid + it * 256;
      int kr = slot >> 5, nq = (slot & 31) * 4;
      float4 f = *(const float4*)(w + (size_t)(ks * 32 + kr) * DIM + n0 + nq);
      ldsBt[(nq + 0) * LDK + kr] = f2bf(f.x);
      ldsBt[(nq + 1) * LDK + kr] = f2bf(f.y);
      ldsBt[(nq + 2) * LDK + kr] = f2bf(f.z);
      ldsBt[(nq + 3) * LDK + kr] = f2bf(f.w);
    }
    asm volatile("s_wait_asynccnt 0" ::: "memory");
    __syncthreads();
    v16bf af[4], bfr[2];
#pragma unroll
    for (int mf = 0; mf < 4; ++mf)
      af[mf] = load_a16x32(&ldsA[(wm * 64 + mf * 16 + lane16) * LDK], hf * 8);
#pragma unroll
    for (int nf = 0; nf < 2; ++nf)
      bfr[nf] = load_b32x16(&ldsBt[(wn * 32 + nf * 16 + lane16) * LDK], hf * 16);
#pragma unroll
    for (int mf = 0; mf < 4; ++mf)
#pragma unroll
      for (int nf = 0; nf < 2; ++nf)
        acc[mf][nf] = wmma_bf16(af[mf], bfr[nf], acc[mf][nf]);
    __syncthreads();
  }
#pragma unroll
  for (int mf = 0; mf < 4; ++mf)
#pragma unroll
    for (int nf = 0; nf < 2; ++nf)
#pragma unroll
      for (int i = 0; i < 8; ++i) {
        int gm = m0 + wm * 64 + mf * 16 + i + hf * 8;
        int gc = n0 + wn * 32 + nf * 16 + lane16;
        out[(size_t)gm * DIM + gc] = acc[mf][nf][i] + bias[gc];
      }
}

extern "C" void kernel_launch(void* const* d_in, const int* in_sizes, int n_in,
                              void* d_out, int out_size, void* d_ws, size_t ws_size,
                              hipStream_t stream) {
  const float* x     = (const float*)d_in[0];  // [2,2048,1024]
  const float* w_qkv = (const float*)d_in[1];  // [1024,3072]
  const float* w_out = (const float*)d_in[2];  // [1024,1024]
  const float* b_out = (const float*)d_in[3];  // [1024]
  float* out = (float*)d_out;                  // [2,2048,1024]

  const size_t QKV_ELEMS = (size_t)BATCH * HEADS * NSEQ * HEAD_DIM;  // 4M
  unsigned short* qb    = (unsigned short*)d_ws;
  unsigned short* kb    = qb + QKV_ELEMS;
  unsigned short* vbt   = kb + QKV_ELEMS;      // V transposed [b,h,64,n]
  unsigned short* attnb = vbt + QKV_ELEMS;     // total 32 MB of ws

  qkv_gemm<<<dim3(3072 / 128, (BATCH * NSEQ) / 128), 256, 0, stream>>>(x, w_qkv, qb, kb, vbt);
  attn_kernel<<<dim3(NSEQ / 64, BATCH * HEADS), 128, 0, stream>>>(qb, kb, vbt, attnb);
  out_gemm<<<dim3(DIM / 128, (BATCH * NSEQ) / 128), 256, 0, stream>>>(attnb, w_out, b_out, out);
}
